// GAU_45191645888690
// MI455X (gfx1250) — matrix operand order repaired and checked by reading, not compiled, not run on hardware
//
#include <hip/hip_runtime.h>
#include <math.h>

// ---------------------------------------------------------------------------
// GAU forward for MI455X (gfx1250, wave32, WMMA bf16 16x16x32, f32 accum)
// Staging uses GLOBAL_LOAD_ASYNC_TO_LDS_B128 (ASYNCcnt) when available.
// ---------------------------------------------------------------------------

typedef __attribute__((ext_vector_type(16))) __bf16 v16bf;
typedef __attribute__((ext_vector_type(8)))  __bf16 v8bf;
typedef __attribute__((ext_vector_type(8)))  float  v8f;
typedef int v4i __attribute__((__vector_size__(16)));   // matches builtin param

#define NTHREADS 256

static constexpr int DIMC  = 1024;
static constexpr int HID   = 2048;
static constexpr int H2    = 4096;   // 2*HID
static constexpr int QKD   = 128;
static constexpr int SEQ   = 4096;
static constexpr int BATCH = 4;
static constexpr int ROWS  = BATCH * SEQ;  // 16384

#define GLOBAL_AS __attribute__((address_space(1)))
#define LDS_AS    __attribute__((address_space(3)))

#if __has_builtin(__builtin_amdgcn_global_load_async_to_lds_b128)
#define USE_ASYNC 1
#else
#define USE_ASYNC 0
#endif

// 16-byte global -> LDS copy (direct-to-LDS async when available)
__device__ __forceinline__ void cp16(__bf16* lds, const __bf16* g) {
#if USE_ASYNC
    __builtin_amdgcn_global_load_async_to_lds_b128(
        (GLOBAL_AS v4i*)g, (LDS_AS v4i*)lds, 0, 0);
#else
    *(uint4*)lds = *(const uint4*)g;
#endif
}

__device__ __forceinline__ void async_wait0() {
#if USE_ASYNC
#if __has_builtin(__builtin_amdgcn_s_wait_asynccnt)
    __builtin_amdgcn_s_wait_asynccnt(0);
#else
    asm volatile("s_wait_asynccnt 0x0" ::: "memory");
#endif
#endif
}

// ---------------- WMMA helpers -------------------------------------------

__device__ __forceinline__ v8f wmma_bf16(v16bf a, v16bf b, v8f c) {
    // D = A(16x32 bf16) * B(32x16 bf16) + C(16x16 f32)
    return __builtin_amdgcn_wmma_f32_16x16x32_bf16(
        /*neg_a=*/false, a, /*neg_b=*/false, b,
        /*c_mod=*/(short)0, c, /*reuse_a=*/false, /*reuse_b=*/false);
}

// A fragment, 16x32 tile, row-major source with row stride ld (elements).
// Lane L: row m = L&15, half = L>>4.  VGPR layout (ISA 7.12.2):
//   elements 0..7  -> k = 8*half + e
//   elements 8..15 -> k = 16 + 8*half + e
__device__ __forceinline__ v16bf ld_afrag(const __bf16* base, int ld, int lane) {
    const int m = lane & 15, h = lane >> 4;
    const __bf16* p = base + m * ld + 8 * h;
    v8bf lo = *(const v8bf*)(p);
    v8bf hi = *(const v8bf*)(p + 16);
    return __builtin_shufflevector(lo, hi, 0,1,2,3,4,5,6,7,8,9,10,11,12,13,14,15);
}

// B fragment, 32x16 tile, from TRANSPOSED storage Bt[n][k] with row stride ld.
// Lane L: col n = L&15, half = L>>4; elements 0..7 -> k = 8*half + e, 8..15 -> k = 16 + 8*half + e.
__device__ __forceinline__ v16bf ld_bfrag(const __bf16* baseT, int ld, int lane) {
    const int n = lane & 15, h = lane >> 4;
    const __bf16* p = baseT + n * ld + 8 * h;
    v8bf lo = *(const v8bf*)(p);
    v8bf hi = *(const v8bf*)(p + 16);
    return __builtin_shufflevector(lo, hi, 0,1,2,3,4,5,6,7,8,9,10,11,12,13,14,15);
}

__device__ __forceinline__ float fast_rcp(float x) {
#if __has_builtin(__builtin_amdgcn_rcpf)
    return __builtin_amdgcn_rcpf(x);
#else
    return 1.0f / x;
#endif
}

__device__ __forceinline__ float silu(float x) {
    return x * fast_rcp(1.0f + __expf(-x));
}

// ---------------- Kernel 1: LayerNorm -> bf16 ----------------------------

__global__ void k_layernorm(const float* __restrict__ x,
                            const float* __restrict__ g,
                            const float* __restrict__ b,
                            __bf16* __restrict__ out) {
    const int row = blockIdx.x;                 // 0..ROWS-1
    const float* xr = x + (size_t)row * DIMC;
    float s = 0.f, s2 = 0.f;
    for (int i = threadIdx.x; i < DIMC; i += NTHREADS) {
        float v = xr[i];
        s += v; s2 += v * v;
    }
    __shared__ float r0[NTHREADS], r1[NTHREADS];
    r0[threadIdx.x] = s; r1[threadIdx.x] = s2;
    __syncthreads();
    for (int off = NTHREADS / 2; off > 0; off >>= 1) {
        if (threadIdx.x < off) {
            r0[threadIdx.x] += r0[threadIdx.x + off];
            r1[threadIdx.x] += r1[threadIdx.x + off];
        }
        __syncthreads();
    }
    const float mu  = r0[0] * (1.0f / DIMC);
    const float var = r1[0] * (1.0f / DIMC) - mu * mu;
    const float inv = rsqrtf(var + 1e-5f);
    __bf16* orow = out + (size_t)row * DIMC;
    for (int i = threadIdx.x; i < DIMC; i += NTHREADS)
        orow[i] = (__bf16)((xr[i] - mu) * inv * g[i] + b[i]);
}

// ---------------- Kernel 2: transpose+convert weight W[K][N] -> Wt[N][K] --

__global__ void k_transpose_bf16(const float* __restrict__ src,
                                 __bf16* __restrict__ dst,
                                 int K, int N) {
    __shared__ float tile[16][17];
    const int k0 = blockIdx.y * 16, n0 = blockIdx.x * 16;
    const int k = k0 + threadIdx.y, n = n0 + threadIdx.x;
    if (k < K && n < N) tile[threadIdx.y][threadIdx.x] = src[(size_t)k * N + n];
    __syncthreads();
    const int kk = k0 + threadIdx.x, nn = n0 + threadIdx.y;
    if (kk < K && nn < N) dst[(size_t)nn * K + kk] = (__bf16)tile[threadIdx.x][threadIdx.y];
}

// ---------------- Kernel 3: hidden GEMM + SiLU, split v/gate --------------
// C[ROWS x H2] = normed[ROWS x 1024] @ Wh^T ; tile 128x128, BK=64.

#define GLDA 72   // 64 + 8 pad, in bf16 elements

__global__ void k_gemm_hidden(const __bf16* __restrict__ A,   // [ROWS x 1024]
                              const __bf16* __restrict__ Bt,  // [4096 x 1024]
                              const float*  __restrict__ bias,
                              __bf16* __restrict__ vout,      // [ROWS x 2048]
                              __bf16* __restrict__ gout) {    // [ROWS x 2048]
    __shared__ __bf16 a_lds[128 * GLDA];
    __shared__ __bf16 b_lds[128 * GLDA];
    const int t = threadIdx.x, lane = t & 31, wave = t >> 5;
    const int wr = wave & 3, wc = wave >> 2;           // 4x2 wave grid
    const int bm = blockIdx.x * 128, bn = blockIdx.y * 128;
    v8f acc[2][4] = {};

    for (int k0 = 0; k0 < DIMC; k0 += 64) {
        __syncthreads();
        #pragma unroll
        for (int it = 0; it < 4; ++it) {
            int chunk = it * NTHREADS + t;             // 0..1023
            int r = chunk >> 3, c = (chunk & 7) * 8;
            cp16(&a_lds[r * GLDA + c], &A [(size_t)(bm + r) * DIMC + k0 + c]);
            cp16(&b_lds[r * GLDA + c], &Bt[(size_t)(bn + r) * DIMC + k0 + c]);
        }
        async_wait0();
        __syncthreads();
        #pragma unroll
        for (int ks = 0; ks < 2; ++ks) {
            v16bf af[2], bfv[4];
            #pragma unroll
            for (int mi = 0; mi < 2; ++mi)
                af[mi] = ld_afrag(&a_lds[(wr * 32 + mi * 16) * GLDA + ks * 32], GLDA, lane);
            #pragma unroll
            for (int ni = 0; ni < 4; ++ni)
                bfv[ni] = ld_bfrag(&b_lds[(wc * 64 + ni * 16) * GLDA + ks * 32], GLDA, lane);
            #pragma unroll
            for (int mi = 0; mi < 2; ++mi)
                #pragma unroll
                for (int ni = 0; ni < 4; ++ni)
                    acc[mi][ni] = wmma_bf16(af[mi], bfv[ni], acc[mi][ni]);
        }
    }
    const int half = lane >> 4, nl = lane & 15;
    #pragma unroll
    for (int mi = 0; mi < 2; ++mi)
        #pragma unroll
        for (int ni = 0; ni < 4; ++ni)
            #pragma unroll
            for (int r = 0; r < 8; ++r) {
                int m = bm + wr * 32 + mi * 16 + r + 8 * half;
                int n = bn + wc * 64 + ni * 16 + nl;
                float c = silu(acc[mi][ni][r] + bias[n]);
                if (n < HID) vout[(size_t)m * HID + n]          = (__bf16)c;
                else         gout[(size_t)m * HID + (n - HID)]  = (__bf16)c;
            }
}

// ---------------- Kernel 4: qk GEMM + SiLU + q/k affine -------------------
// C[ROWS x 128] = normed @ Wqk^T ; one 128-col block.

__global__ void k_gemm_qk(const __bf16* __restrict__ A,    // [ROWS x 1024]
                          const __bf16* __restrict__ Bt,   // [128 x 1024]
                          const float*  __restrict__ bias, // [128]
                          const float*  __restrict__ gam,  // [2 x 128]
                          const float*  __restrict__ bet,  // [2 x 128]
                          __bf16* __restrict__ qout,       // [ROWS x 128]
                          __bf16* __restrict__ kout) {     // [ROWS x 128]
    __shared__ __bf16 a_lds[128 * GLDA];
    __shared__ __bf16 b_lds[128 * GLDA];
    const int t = threadIdx.x, lane = t & 31, wave = t >> 5;
    const int wr = wave & 3, wc = wave >> 2;
    const int bm = blockIdx.x * 128;
    v8f acc[2][4] = {};

    for (int k0 = 0; k0 < DIMC; k0 += 64) {
        __syncthreads();
        #pragma unroll
        for (int it = 0; it < 4; ++it) {
            int chunk = it * NTHREADS + t;
            int r = chunk >> 3, c = (chunk & 7) * 8;
            cp16(&a_lds[r * GLDA + c], &A [(size_t)(bm + r) * DIMC + k0 + c]);
            cp16(&b_lds[r * GLDA + c], &Bt[(size_t)r        * DIMC + k0 + c]);
        }
        async_wait0();
        __syncthreads();
        #pragma unroll
        for (int ks = 0; ks < 2; ++ks) {
            v16bf af[2], bfv[4];
            #pragma unroll
            for (int mi = 0; mi < 2; ++mi)
                af[mi] = ld_afrag(&a_lds[(wr * 32 + mi * 16) * GLDA + ks * 32], GLDA, lane);
            #pragma unroll
            for (int ni = 0; ni < 4; ++ni)
                bfv[ni] = ld_bfrag(&b_lds[(wc * 64 + ni * 16) * GLDA + ks * 32], GLDA, lane);
            #pragma unroll
            for (int mi = 0; mi < 2; ++mi)
                #pragma unroll
                for (int ni = 0; ni < 4; ++ni)
                    acc[mi][ni] = wmma_bf16(af[mi], bfv[ni], acc[mi][ni]);
        }
    }
    const int half = lane >> 4, nl = lane & 15;
    #pragma unroll
    for (int mi = 0; mi < 2; ++mi)
        #pragma unroll
        for (int ni = 0; ni < 4; ++ni)
            #pragma unroll
            for (int r = 0; r < 8; ++r) {
                int m = bm + wr * 32 + mi * 16 + r + 8 * half;
                int n = wc * 64 + ni * 16 + nl;                 // 0..127
                float qk = silu(acc[mi][ni][r] + bias[n]);
                qout[(size_t)m * QKD + n] = (__bf16)(qk * gam[n]       + bet[n]);
                kout[(size_t)m * QKD + n] = (__bf16)(qk * gam[QKD + n] + bet[QKD + n]);
            }
}

// ---------------- Kernel 5: causal relu^2 attention + gate ---------------
// grid: x = H-slice (16 of 128 cols), y = 64-row block (64), z = batch (4)

#define PLD 72    // 64+8 pad
#define QLD 136   // 128+8 pad

__global__ void k_attn(const __bf16* __restrict__ qb,    // [ROWS x 128]
                       const __bf16* __restrict__ kb,    // [ROWS x 128]
                       const __bf16* __restrict__ vb,    // [ROWS x 2048]
                       const __bf16* __restrict__ gateb, // [ROWS x 2048]
                       __bf16* __restrict__ gated) {     // [ROWS x 2048]
    const int hb = blockIdx.x, ib = blockIdx.y, bz = blockIdx.z;
    __shared__ __bf16 q_lds [64 * QLD];
    __shared__ __bf16 k_lds [64 * QLD];
    __shared__ __bf16 p_lds [64 * PLD];
    __shared__ __bf16 vt_lds[128 * PLD];
    const int t = threadIdx.x, lane = t & 31, wave = t >> 5;
    const int half = lane >> 4, nl = lane & 15;
    const size_t rowbase = (size_t)bz * SEQ + (size_t)ib * 64;

    // stage q tile [64 x 128] once
    #pragma unroll
    for (int it = 0; it < 4; ++it) {
        int chunk = it * NTHREADS + t;                 // 0..1023
        int r = chunk >> 4, c = (chunk & 15) * 8;
        cp16(&q_lds[r * QLD + c], &qb[(rowbase + r) * QKD + c]);
    }

    // output accumulators: wave = 4x2 grid over 64 x 128 tile
    const int m2 = wave & 3;                 // 16-row tile
    const int nb2 = (wave >> 2) * 64;        // 64-col half
    v8f acc[4] = {};

    // S-stage assignment: wave -> m-tile (wave&3), two n-tiles
    const int sm = wave & 3;
    const int sn0 = (wave >> 2) * 2;

    async_wait0();
    __syncthreads();

    for (int jb = 0; jb <= ib; ++jb) {
        const size_t jrow = (size_t)bz * SEQ + (size_t)jb * 64;
        // stage k tile [64 x 128]
        #pragma unroll
        for (int it = 0; it < 4; ++it) {
            int chunk = it * NTHREADS + t;
            int r = chunk >> 4, c = (chunk & 15) * 8;
            cp16(&k_lds[r * QLD + c], &kb[(jrow + r) * QKD + c]);
        }
        // stage transposed V slice: vt[h][j] = v[jrow+j][hb*128+h]
        #pragma unroll
        for (int it = 0; it < 32; ++it) {
            int idx = it * NTHREADS + t;               // 0..8191
            int j = idx >> 7, h = idx & 127;
            vt_lds[h * PLD + j] = vb[(jrow + j) * (size_t)HID + hb * 128 + h];
        }
        async_wait0();
        __syncthreads();

        // S = q @ k^T  (64x64), this wave: tiles (sm, sn0..sn0+1)
        v8f sacc[2] = {};
        #pragma unroll
        for (int ks = 0; ks < 4; ++ks) {
            v16bf aq = ld_afrag(&q_lds[sm * 16 * QLD + ks * 32], QLD, lane);
            #pragma unroll
            for (int ni = 0; ni < 2; ++ni) {
                v16bf bk = ld_bfrag(&k_lds[(sn0 + ni) * 16 * QLD + ks * 32], QLD, lane);
                sacc[ni] = wmma_bf16(aq, bk, sacc[ni]);
            }
        }
        // scale 1/(i+1), relu^2, causal mask; write P (bf16) to LDS
        #pragma unroll
        for (int ni = 0; ni < 2; ++ni)
            #pragma unroll
            for (int r = 0; r < 8; ++r) {
                int il = sm * 16 + r + 8 * half;
                int jl = (sn0 + ni) * 16 + nl;
                int ig = ib * 64 + il, jg = jb * 64 + jl;
                float s = sacc[ni][r] * fast_rcp((float)(ig + 1));
                s = fmaxf(s, 0.0f);
                s = s * s;
                if (jg > ig) s = 0.0f;
                p_lds[il * PLD + jl] = (__bf16)s;
            }
        __syncthreads();

        // acc += P (64x64) @ Vt-slice (64 x 128)
        #pragma unroll
        for (int ks = 0; ks < 2; ++ks) {
            v16bf ap = ld_afrag(&p_lds[m2 * 16 * PLD + ks * 32], PLD, lane);
            #pragma unroll
            for (int nt = 0; nt < 4; ++nt) {
                v16bf bv = ld_bfrag(&vt_lds[(nb2 + nt * 16) * PLD + ks * 32], PLD, lane);
                acc[nt] = wmma_bf16(ap, bv, acc[nt]);
            }
        }
        __syncthreads();   // protect k_lds / vt_lds / p_lds before next jb
    }

    // epilogue: multiply by gate, store bf16
    #pragma unroll
    for (int nt = 0; nt < 4; ++nt)
        #pragma unroll
        for (int r = 0; r < 8; ++r) {
            int m = m2 * 16 + r + 8 * half;
            int h = nb2 + nt * 16 + nl;
            size_t row = rowbase + m;
            size_t col = (size_t)hb * 128 + h;
            float g = (float)gateb[row * HID + col];
            gated[row * HID + col] = (__bf16)(acc[nt][r] * g);
        }
}

// ---------------- Kernel 6: output GEMM + bias + residual ----------------
// out[ROWS x 1024] = gated[ROWS x 2048] @ Wout^T + b_out + x

__global__ void k_gemm_out(const __bf16* __restrict__ A,   // [ROWS x 2048]
                           const __bf16* __restrict__ Bt,  // [1024 x 2048]
                           const float*  __restrict__ bias,
                           const float*  __restrict__ xin,
                           float* __restrict__ out) {
    __shared__ __bf16 a_lds[128 * GLDA];
    __shared__ __bf16 b_lds[128 * GLDA];
    const int t = threadIdx.x, lane = t & 31, wave = t >> 5;
    const int wr = wave & 3, wc = wave >> 2;
    const int bm = blockIdx.x * 128, bn = blockIdx.y * 128;
    v8f acc[2][4] = {};

    for (int k0 = 0; k0 < HID; k0 += 64) {
        __syncthreads();
        #pragma unroll
        for (int it = 0; it < 4; ++it) {
            int chunk = it * NTHREADS + t;
            int r = chunk >> 3, c = (chunk & 7) * 8;
            cp16(&a_lds[r * GLDA + c], &A [(size_t)(bm + r) * HID + k0 + c]);
            cp16(&b_lds[r * GLDA + c], &Bt[(size_t)(bn + r) * HID + k0 + c]);
        }
        async_wait0();
        __syncthreads();
        #pragma unroll
        for (int ks = 0; ks < 2; ++ks) {
            v16bf af[2], bfv[4];
            #pragma unroll
            for (int mi = 0; mi < 2; ++mi)
                af[mi] = ld_afrag(&a_lds[(wr * 32 + mi * 16) * GLDA + ks * 32], GLDA, lane);
            #pragma unroll
            for (int ni = 0; ni < 4; ++ni)
                bfv[ni] = ld_bfrag(&b_lds[(wc * 64 + ni * 16) * GLDA + ks * 32], GLDA, lane);
            #pragma unroll
            for (int mi = 0; mi < 2; ++mi)
                #pragma unroll
                for (int ni = 0; ni < 4; ++ni)
                    acc[mi][ni] = wmma_bf16(af[mi], bfv[ni], acc[mi][ni]);
        }
    }
    const int half = lane >> 4, nl = lane & 15;
    #pragma unroll
    for (int mi = 0; mi < 2; ++mi)
        #pragma unroll
        for (int ni = 0; ni < 4; ++ni)
            #pragma unroll
            for (int r = 0; r < 8; ++r) {
                int m = bm + wr * 32 + mi * 16 + r + 8 * half;
                int n = bn + wc * 64 + ni * 16 + nl;
                out[(size_t)m * DIMC + n] =
                    acc[mi][ni][r] + bias[n] + xin[(size_t)m * DIMC + n];
            }
}

// ---------------- Host launcher ------------------------------------------

extern "C" void kernel_launch(void* const* d_in, const int* in_sizes, int n_in,
                              void* d_out, int out_size, void* d_ws, size_t ws_size,
                              hipStream_t stream) {
    const float* x        = (const float*)d_in[0];
    const float* ln_g     = (const float*)d_in[1];
    const float* ln_b     = (const float*)d_in[2];
    const float* W_hidden = (const float*)d_in[3];
    const float* b_hidden = (const float*)d_in[4];
    const float* W_qk     = (const float*)d_in[5];
    const float* b_qk     = (const float*)d_in[6];
    const float* os_gamma = (const float*)d_in[7];
    const float* os_beta  = (const float*)d_in[8];
    const float* W_out    = (const float*)d_in[9];
    const float* b_out    = (const float*)d_in[10];
    float* out = (float*)d_out;

    char* ws = (char*)d_ws;
    size_t off = 0;
    auto carve = [&](size_t bytes) -> char* {
        char* p = ws + off;
        off += (bytes + 255) & ~(size_t)255;
        return p;
    };
    __bf16* normed = (__bf16*)carve((size_t)ROWS * DIMC * 2);
    __bf16* wh_t   = (__bf16*)carve((size_t)H2   * DIMC * 2);
    __bf16* wqk_t  = (__bf16*)carve((size_t)QKD  * DIMC * 2);
    __bf16* wout_t = (__bf16*)carve((size_t)DIMC * HID  * 2);
    __bf16* v_buf  = (__bf16*)carve((size_t)ROWS * HID  * 2);
    __bf16* g_buf  = (__bf16*)carve((size_t)ROWS * HID  * 2);
    __bf16* q_buf  = (__bf16*)carve((size_t)ROWS * QKD  * 2);
    __bf16* k_buf  = (__bf16*)carve((size_t)ROWS * QKD  * 2);
    __bf16* gated  = (__bf16*)carve((size_t)ROWS * HID  * 2);
    (void)ws_size; (void)in_sizes; (void)n_in; (void)out_size;

    // 1. LayerNorm -> bf16
    k_layernorm<<<ROWS, NTHREADS, 0, stream>>>(x, ln_g, ln_b, normed);

    // 2. Weight transposes (f32 -> bf16, [K x N] -> [N x K])
    k_transpose_bf16<<<dim3(H2 / 16,  DIMC / 16), dim3(16, 16), 0, stream>>>(W_hidden, wh_t,  DIMC, H2);
    k_transpose_bf16<<<dim3(QKD / 16, DIMC / 16), dim3(16, 16), 0, stream>>>(W_qk,     wqk_t, DIMC, QKD);
    k_transpose_bf16<<<dim3(DIMC / 16, HID / 16), dim3(16, 16), 0, stream>>>(W_out,    wout_t, HID, DIMC);

    // 3. hidden GEMM + SiLU -> v, gate
    k_gemm_hidden<<<dim3(ROWS / 128, H2 / 128), NTHREADS, 0, stream>>>(
        normed, wh_t, b_hidden, v_buf, g_buf);

    // 4. qk GEMM + SiLU + affine -> q, k
    k_gemm_qk<<<dim3(ROWS / 128, 1), NTHREADS, 0, stream>>>(
        normed, wqk_t, b_qk, os_gamma, os_beta, q_buf, k_buf);

    // 5. causal relu^2 attention + gate
    k_attn<<<dim3(HID / 128, SEQ / 64, BATCH), NTHREADS, 0, stream>>>(
        q_buf, k_buf, v_buf, g_buf, gated);

    // 6. output GEMM + bias + residual
    k_gemm_out<<<dim3(ROWS / 128, DIMC / 128), NTHREADS, 0, stream>>>(
        gated, wout_t, b_out, x, out);
}
